// ComplexMultiHeadAttention_14379550507634
// MI455X (gfx1250) — compile-verified, matrix operand order checked by reading, not run on hardware
//
#include <hip/hip_runtime.h>

// ---------------- CDNA5 / gfx1250 WMMA complex multi-head attention ----------
// B=2, L=2048, D=1024, H=16, HD=64. All GEMMs via v_wmma_f32_16x16x32_f16.
// Pipeline: f32->f16 convert -> QKV complex GEMM + fused RoPE -> flash attention
// (4-wave workgroups, K/V staged in LDS with global_load_async_to_lds_b128,
// online softmax, P@[vr|vi]) -> complex output projection + bias (f32 out).

static constexpr int B_  = 2;
static constexpr int L_  = 2048;
static constexpr int D_  = 1024;
static constexpr int H_  = 16;
static constexpr int HD_ = 64;

typedef __attribute__((ext_vector_type(16))) _Float16 v16h;
typedef __attribute__((ext_vector_type(8)))  _Float16 v8h;
typedef __attribute__((ext_vector_type(8)))  float    v8f;

#define DEVFN __device__ __forceinline__

DEVFN v8f wmma32(v16h a, v16h b, v8f c) {
  // D = A(16x32 f16) * B(32x16 f16) + C(16x16 f32)
  return __builtin_amdgcn_wmma_f32_16x16x32_f16(
      /*neg_a=*/false, a, /*neg_b=*/false, b,
      /*c_mod=*/(short)0, c, /*reuse_a=*/false, /*reuse_b=*/false);
}

// Load a 16x32 f16 tile in WMMA A/B register layout.
// Lane l (<16): row l, K chunks {0..7, 16..23}; lane l+16: row l, K {8..15, 24..31}.
// Requires base 16B-aligned and stride a multiple of 8 elements.
DEVFN v16h load_tile16(const _Float16* __restrict__ base, int stride, int lane) {
  const int r  = lane & 15;
  const int hi = lane >> 4;
  const _Float16* p = base + (size_t)r * stride + hi * 8;
  v8h lo = *(const v8h*)(p);
  v8h hh = *(const v8h*)(p + 16);
  v16h out;
#pragma unroll
  for (int i = 0; i < 8; ++i) { out[i] = lo[i]; out[i + 8] = hh[i]; }
  return out;
}

// ---- CDNA5 async global->LDS copy (ASYNCcnt path) ---------------------------
DEVFN unsigned lds_off_of(const void* p) {
  // Low 32 bits of a generic pointer into LDS == DS-relative byte offset.
  return (unsigned)(unsigned long long)p;
}
DEVFN void async_copy_b128(unsigned lds_byte_off, const _Float16* gsrc) {
  asm volatile("global_load_async_to_lds_b128 %0, %1, off"
               :: "v"(lds_byte_off), "v"((unsigned long long)gsrc)
               : "memory");
}
DEVFN void wait_async0() {
  asm volatile("s_wait_asynccnt 0" ::: "memory");
}

// ---------------------------------------------------------------- convert ----
__global__ void cvt_f16_kernel(const float* __restrict__ src,
                               _Float16* __restrict__ dst, int n, float scale) {
  int i = blockIdx.x * blockDim.x + threadIdx.x;
  if (i < n) dst[i] = (_Float16)(src[i] * scale);
}

// ------------------------------------------------- QKV projection + RoPE ----
// One wave computes 16 tokens x one head (64 feats) of real+imag.
// real = xr@wr.T + xi@(-wi).T ; imag = xr@wi.T + xi@wr.T
// mode 0/1: RoPE then store [b,h,l,128] (real|imag). mode 2: store V^T [b,h,128,l].
__global__ __launch_bounds__(32) void qkv_rope_kernel(
    const _Float16* __restrict__ xr, const _Float16* __restrict__ xi,
    const _Float16* __restrict__ wr, const _Float16* __restrict__ wi,
    const _Float16* __restrict__ wineg,
    _Float16* __restrict__ out, int mode) {
  const int lane = threadIdx.x;
  const int hi   = lane >> 4;
  const int ln   = lane & 15;
  const int m0   = blockIdx.x * 16;   // token row in [0, B*L)
  const int h    = blockIdx.y;        // head

  v8f accr[4] = {}, acci[4] = {};

  for (int k0 = 0; k0 < D_; k0 += 32) {
    __builtin_prefetch(xr + (size_t)m0 * D_ + k0 + 128, 0, 0);
    __builtin_prefetch(xi + (size_t)m0 * D_ + k0 + 128, 0, 0);
    v16h axr = load_tile16(xr + (size_t)m0 * D_ + k0, D_, lane);
    v16h axi = load_tile16(xi + (size_t)m0 * D_ + k0, D_, lane);
#pragma unroll
    for (int t = 0; t < 4; ++t) {
      const int n0 = h * HD_ + t * 16;
      v16h bwr = load_tile16(wr    + (size_t)n0 * D_ + k0, D_, lane);
      v16h bwi = load_tile16(wi    + (size_t)n0 * D_ + k0, D_, lane);
      v16h bwn = load_tile16(wineg + (size_t)n0 * D_ + k0, D_, lane);
      accr[t] = wmma32(axr, bwr, accr[t]);
      accr[t] = wmma32(axi, bwn, accr[t]);
      acci[t] = wmma32(axr, bwi, acci[t]);
      acci[t] = wmma32(axi, bwr, acci[t]);
    }
  }

  if (mode < 2) {
    // RoPE on accumulators: feature f=t*16+ln; pairs (t, t+2) are (d, d+32).
#pragma unroll
    for (int t = 0; t < 2; ++t) {
      const float invf = __powf(10000.f, -(float)(t * 16 + ln) * (1.0f / 32.0f));
#pragma unroll
      for (int j = 0; j < 8; ++j) {
        const int   m   = m0 + j + hi * 8;
        const float pos = (float)(m & (L_ - 1));
        float s, c;
        __sincosf(pos * invf, &s, &c);
        float rl = accr[t][j], rh = accr[t + 2][j];
        accr[t][j]     = rl * c - rh * s;
        accr[t + 2][j] = rh * c + rl * s;
        float il = acci[t][j], ih = acci[t + 2][j];
        acci[t][j]     = il * c - ih * s;
        acci[t + 2][j] = ih * c + il * s;
      }
    }
    // Store to cat layout [b, h, l, 128] (real 0..63 | imag 64..127)
#pragma unroll
    for (int t = 0; t < 4; ++t) {
#pragma unroll
      for (int j = 0; j < 8; ++j) {
        const int m = m0 + j + hi * 8;
        const int b = m >> 11, l = m & (L_ - 1);
        const int f = t * 16 + ln;
        const size_t base = (((size_t)(b * H_ + h)) * L_ + l) * 128;
        out[base + f]      = (_Float16)accr[t][j];
        out[base + 64 + f] = (_Float16)acci[t][j];
      }
    }
  } else {
    // V transposed: [b, h, c(0..127), l]
#pragma unroll
    for (int t = 0; t < 4; ++t) {
#pragma unroll
      for (int j = 0; j < 8; ++j) {
        const int m = m0 + j + hi * 8;
        const int b = m >> 11, l = m & (L_ - 1);
        const int f = t * 16 + ln;
        const size_t base = ((size_t)(b * H_ + h)) * 128;
        out[(base + f) * L_ + l]      = (_Float16)accr[t][j];
        out[(base + 64 + f) * L_ + l] = (_Float16)acci[t][j];
      }
    }
  }
}

// ------------------------------------------------------ flash attention -----
// 4 waves / block = 64 queries of one (b,h). Keys processed 32 at a time.
// Each 32-key step: the block cooperatively async-copies K (32x128, contiguous
// 8KB) and V^T (128 rows x 64B) into LDS, waits ASYNCcnt, barriers, then all
// waves run QK^T WMMAs + online softmax + P@V WMMAs entirely from LDS.
__global__ __launch_bounds__(128) void attn_kernel(
    const _Float16* __restrict__ Qc, const _Float16* __restrict__ Kc,
    const _Float16* __restrict__ Vt,
    _Float16* __restrict__ Or, _Float16* __restrict__ Oi) {
  __shared__ __align__(16) _Float16 Kl[32 * 128];     // [key][ch]   8KB
  __shared__ __align__(16) _Float16 Vl[128 * 32];     // [feat][key] 8KB
  __shared__ __align__(16) _Float16 Pl[4][16 * 32];   // per-wave P  4KB

  const int tid  = threadIdx.x;
  const int wave = tid >> 5;
  const int lane = tid & 31;
  const int hi   = lane >> 4;
  const int ln   = lane & 15;
  const int qblk = blockIdx.x * 64;
  const int q0   = qblk + wave * 16;
  const int bh   = blockIdx.y;

  const _Float16* Qb = Qc + (size_t)bh * L_ * 128;
  const _Float16* Kb = Kc + (size_t)bh * L_ * 128;
  const _Float16* Vb = Vt + (size_t)bh * 128 * L_;

  const unsigned klo = lds_off_of(&Kl[0]);
  const unsigned vlo = lds_off_of(&Vl[0]);

  v16h qa[4];
#pragma unroll
  for (int j = 0; j < 4; ++j)
    qa[j] = load_tile16(Qb + (size_t)q0 * 128 + j * 32, 128, lane);

  v8f   o[8] = {};
  float rmax[8], rsum[8];
#pragma unroll
  for (int j = 0; j < 8; ++j) { rmax[j] = -1e30f; rsum[j] = 0.f; }

  const int kend = qblk + 48;  // last q-tile of the block sets the causal reach
  for (int k0 = 0; k0 <= kend; k0 += 32) {
    // ---- cooperative async global->LDS stage (ASYNCcnt) ----
#pragma unroll
    for (int i = 0; i < 4; ++i) {
      const int c = tid + i * 128;            // 512 x 16B chunks each
      // K block: contiguous 8KB at Kb + k0*128 elements
      async_copy_b128(klo + c * 16, Kb + (size_t)k0 * 128 + c * 8);
      // V block: 128 feature rows, 64B (32 keys) per row, stride L_
      const int feat = c >> 2, sub = c & 3;
      async_copy_b128(vlo + c * 16, Vb + (size_t)feat * L_ + k0 + sub * 8);
    }
    wait_async0();
    __syncthreads();

    // ---- scores from LDS ----
    v8f s0 = {}, s1 = {};
#pragma unroll
    for (int j = 0; j < 4; ++j) {
      v16h kb0 = load_tile16(&Kl[0] + j * 32, 128, lane);
      v16h kb1 = load_tile16(&Kl[0] + 16 * 128 + j * 32, 128, lane);
      s0 = wmma32(qa[j], kb0, s0);
      s1 = wmma32(qa[j], kb1, s1);
    }

    const float scale = 0.125f;  // 1/sqrt(64)
    float p0[8], p1[8];
#pragma unroll
    for (int j = 0; j < 8; ++j) {
      const int q = q0 + j + hi * 8;
      float a0 = s0[j] * scale; if (k0 + ln > q)      a0 = -1e30f;
      float a1 = s1[j] * scale; if (k0 + 16 + ln > q) a1 = -1e30f;
      float bm = fmaxf(a0, a1);
      bm = fmaxf(bm, __shfl_xor(bm, 1));
      bm = fmaxf(bm, __shfl_xor(bm, 2));
      bm = fmaxf(bm, __shfl_xor(bm, 4));
      bm = fmaxf(bm, __shfl_xor(bm, 8));
      const float mnew = fmaxf(rmax[j], bm);
      const float corr = __expf(rmax[j] - mnew);
      rmax[j] = mnew;
      const float e0 = __expf(a0 - mnew);
      const float e1 = __expf(a1 - mnew);
      float ps = e0 + e1;
      ps += __shfl_xor(ps, 1);
      ps += __shfl_xor(ps, 2);
      ps += __shfl_xor(ps, 4);
      ps += __shfl_xor(ps, 8);
      rsum[j] = rsum[j] * corr + ps;
#pragma unroll
      for (int t = 0; t < 8; ++t) o[t][j] *= corr;
      p0[j] = e0; p1[j] = e1;
    }

    // C-layout P -> per-wave LDS [16][32] -> reload in A layout (f16).
    // DS ops from one wave are in-order, so no barrier needed here.
#pragma unroll
    for (int j = 0; j < 8; ++j) {
      Pl[wave][(j + hi * 8) * 32 + ln]      = (_Float16)p0[j];
      Pl[wave][(j + hi * 8) * 32 + 16 + ln] = (_Float16)p1[j];
    }
    v16h pa = load_tile16(&Pl[wave][0], 32, lane);

    // ---- P @ V from LDS ----
#pragma unroll
    for (int t = 0; t < 8; ++t) {
      v16h vb = load_tile16(&Vl[0] + (t * 16) * 32, 32, lane);
      o[t] = wmma32(pa, vb, o[t]);
    }
    __syncthreads();  // all waves done with Kl/Vl before next stage overwrites
  }

  const int b = bh >> 4, h = bh & 15;
#pragma unroll
  for (int j = 0; j < 8; ++j) {
    const float inv = 1.0f / rsum[j];
    const int m = q0 + j + hi * 8;
    const size_t row = ((size_t)b * L_ + m) * D_;
#pragma unroll
    for (int t = 0; t < 8; ++t) {
      const int f = t * 16 + ln;
      _Float16* dst = (t < 4) ? Or : Oi;
      dst[row + h * HD_ + (f & 63)] = (_Float16)(o[t][j] * inv);
    }
  }
}

// --------------------------------------------------- output projection ------
__global__ __launch_bounds__(32) void oproj_kernel(
    const _Float16* __restrict__ Or, const _Float16* __restrict__ Oi,
    const _Float16* __restrict__ wr, const _Float16* __restrict__ wi,
    const _Float16* __restrict__ wineg,
    const float* __restrict__ br, const float* __restrict__ bi,
    float* __restrict__ outr, float* __restrict__ outi) {
  const int lane = threadIdx.x;
  const int hi   = lane >> 4;
  const int ln   = lane & 15;
  const int m0   = blockIdx.x * 16;
  const int n0   = blockIdx.y * 16;

  v8f ar = {}, ai = {};
  for (int k0 = 0; k0 < D_; k0 += 32) {
    __builtin_prefetch(Or + (size_t)m0 * D_ + k0 + 128, 0, 0);
    v16h aor = load_tile16(Or + (size_t)m0 * D_ + k0, D_, lane);
    v16h aoi = load_tile16(Oi + (size_t)m0 * D_ + k0, D_, lane);
    v16h bwr = load_tile16(wr    + (size_t)n0 * D_ + k0, D_, lane);
    v16h bwi = load_tile16(wi    + (size_t)n0 * D_ + k0, D_, lane);
    v16h bwn = load_tile16(wineg + (size_t)n0 * D_ + k0, D_, lane);
    ar = wmma32(aor, bwr, ar);
    ar = wmma32(aoi, bwn, ar);
    ai = wmma32(aor, bwi, ai);
    ai = wmma32(aoi, bwr, ai);
  }

  const int   n   = n0 + ln;
  const float cbr = br[n], cbi = bi[n];
#pragma unroll
  for (int j = 0; j < 8; ++j) {
    const int m = m0 + j + hi * 8;
    outr[(size_t)m * D_ + n] = ar[j] + cbr;
    outi[(size_t)m * D_ + n] = ai[j] + cbi;
  }
}

// -------------------------------------------------------------- launcher ----
extern "C" void kernel_launch(void* const* d_in, const int* in_sizes, int n_in,
                              void* d_out, int out_size, void* d_ws, size_t ws_size,
                              hipStream_t stream) {
  (void)in_sizes; (void)n_in; (void)out_size; (void)ws_size;

  const float* x_real = (const float*)d_in[0];
  const float* x_imag = (const float*)d_in[1];
  const float* wq_r   = (const float*)d_in[2];
  const float* wq_i   = (const float*)d_in[3];
  const float* wk_r   = (const float*)d_in[4];
  const float* wk_i   = (const float*)d_in[5];
  const float* wv_r   = (const float*)d_in[6];
  const float* wv_i   = (const float*)d_in[7];
  const float* wo_r   = (const float*)d_in[8];
  const float* wo_i   = (const float*)d_in[9];
  const float* bo_r   = (const float*)d_in[10];
  const float* bo_i   = (const float*)d_in[11];

  const size_t NX = (size_t)B_ * L_ * D_;  // 4,194,304
  const size_t NW = (size_t)D_ * D_;       // 1,048,576
  const size_t NC = (size_t)B_ * H_ * L_ * 128;

  char*  ws  = (char*)d_ws;
  size_t off = 0;
  auto alloc = [&](size_t bytes) -> _Float16* {
    char* p = ws + off;
    off += (bytes + 255) & ~(size_t)255;
    return (_Float16*)p;
  };

  _Float16* xr_h   = alloc(NX * 2);
  _Float16* xi_h   = alloc(NX * 2);
  _Float16* wqr_h  = alloc(NW * 2);
  _Float16* wqi_h  = alloc(NW * 2);
  _Float16* wqin_h = alloc(NW * 2);
  _Float16* wkr_h  = alloc(NW * 2);
  _Float16* wki_h  = alloc(NW * 2);
  _Float16* wkin_h = alloc(NW * 2);
  _Float16* wvr_h  = alloc(NW * 2);
  _Float16* wvi_h  = alloc(NW * 2);
  _Float16* wvin_h = alloc(NW * 2);
  _Float16* wor_h  = alloc(NW * 2);
  _Float16* woi_h  = alloc(NW * 2);
  _Float16* woin_h = alloc(NW * 2);
  _Float16* Qcat   = alloc(NC * 2);
  _Float16* Kcat   = alloc(NC * 2);
  _Float16* Vt     = alloc(NC * 2);
  _Float16* Or_h   = alloc(NX * 2);
  _Float16* Oi_h   = alloc(NX * 2);

  auto cvt = [&](const float* s, _Float16* d, size_t n, float sc) {
    cvt_f16_kernel<<<dim3((unsigned)((n + 255) / 256)), dim3(256), 0, stream>>>(
        s, d, (int)n, sc);
  };

  cvt(x_real, xr_h, NX, 1.f);
  cvt(x_imag, xi_h, NX, 1.f);
  cvt(wq_r, wqr_h, NW, 1.f); cvt(wq_i, wqi_h, NW, 1.f); cvt(wq_i, wqin_h, NW, -1.f);
  cvt(wk_r, wkr_h, NW, 1.f); cvt(wk_i, wki_h, NW, 1.f); cvt(wk_i, wkin_h, NW, -1.f);
  cvt(wv_r, wvr_h, NW, 1.f); cvt(wv_i, wvi_h, NW, 1.f); cvt(wv_i, wvin_h, NW, -1.f);
  cvt(wo_r, wor_h, NW, 1.f); cvt(wo_i, woi_h, NW, 1.f); cvt(wo_i, woin_h, NW, -1.f);

  const dim3 gq((B_ * L_) / 16, H_);
  qkv_rope_kernel<<<gq, 32, 0, stream>>>(xr_h, xi_h, wqr_h, wqi_h, wqin_h, Qcat, 0);
  qkv_rope_kernel<<<gq, 32, 0, stream>>>(xr_h, xi_h, wkr_h, wki_h, wkin_h, Kcat, 1);
  qkv_rope_kernel<<<gq, 32, 0, stream>>>(xr_h, xi_h, wvr_h, wvi_h, wvin_h, Vt, 2);

  attn_kernel<<<dim3(L_ / 64, B_ * H_), 128, 0, stream>>>(Qcat, Kcat, Vt, Or_h, Oi_h);

  float* out_r = (float*)d_out;
  float* out_i = out_r + NX;
  oproj_kernel<<<dim3((B_ * L_) / 16, D_ / 16), 32, 0, stream>>>(
      Or_h, Oi_h, wor_h, woi_h, woin_h, bo_r, bo_i, out_r, out_i);
}
